// TemporalEncoderLayer_15985868276500
// MI455X (gfx1250) — matrix operand order repaired and checked by reading, not compiled, not run on hardware
//
#include <hip/hip_runtime.h>
#include <hip/hip_bf16.h>

typedef __bf16 bf16;
typedef __attribute__((ext_vector_type(16))) __bf16 v16bf;
typedef __attribute__((ext_vector_type(8)))  float   v8f;

#define F_DIM 128
#define N_DIM 512
#define MDL   512
#define HEADS 8
#define E_DIM 64
#define M_FEAT 128
#define ROWS  (F_DIM * N_DIM)   // 65536

// ---------------------------------------------------------------------------
// CDNA5 async global->LDS copy (ASYNCcnt-tracked, bypasses VGPRs)
// LDS operand: generic shared pointer's low 32 bits are the LDS byte address.
// ---------------------------------------------------------------------------
__device__ __forceinline__ void async_copy_b128(void* lds_dst, const void* gsrc) {
  unsigned ldsa = (unsigned)(uintptr_t)lds_dst;
  asm volatile("global_load_async_to_lds_b128 %0, %1, off"
               :: "v"(ldsa), "v"(gsrc) : "memory");
}
__device__ __forceinline__ void wait_async0() {
  asm volatile("s_wait_asynccnt 0" ::: "memory");
}

// ---------------------------------------------------------------------------
// WMMA helpers (CDNA5 16x16x32 bf16, f32 accumulate)
// A layout (16-bit 16x32): lane l(0..15)=row M; element i -> k=(i/8)*16+hi*8+(i%8)
// B layout (16-bit 32x16): lane = col N; element i -> k=(i/8)*16+hi*8+(i%8)
// C/D layout: vgpr r -> m = r + 8*hi, lane -> n
// ---------------------------------------------------------------------------
__device__ __forceinline__ v8f wmma_bf16(v16bf a, v16bf b, v8f c) {
  return __builtin_amdgcn_wmma_f32_16x16x32_bf16(false, a, false, b, (short)0, c,
                                                 false, false);
}
__device__ __forceinline__ int kA_idx(int i, int hi) { return ((i >> 3) << 4) + (hi << 3) + (i & 7); }
__device__ __forceinline__ int kB_idx(int i, int hi) { return ((i >> 3) << 4) + (hi << 3) + (i & 7); }

__device__ __forceinline__ v16bf load_a_bf16(const bf16* src, int ld, int lane) {
  int l = lane & 15, hi = lane >> 4;
  v16bf a;
#pragma unroll
  for (int i = 0; i < 16; ++i) a[i] = src[l * ld + kA_idx(i, hi)];
  return a;
}
__device__ __forceinline__ v16bf load_a_f32(const float* src, int ld, int lane) {
  int l = lane & 15, hi = lane >> 4;
  v16bf a;
#pragma unroll
  for (int i = 0; i < 16; ++i) a[i] = (bf16)src[l * ld + kA_idx(i, hi)];
  return a;
}
__device__ __forceinline__ v16bf load_at_f32(const float* src, int ld, int lane) {
  int l = lane & 15, hi = lane >> 4;
  v16bf a;
#pragma unroll
  for (int i = 0; i < 16; ++i) a[i] = (bf16)src[kA_idx(i, hi) * ld + l];
  return a;
}
__device__ __forceinline__ v16bf load_b_rm(const bf16* src, int ld, int lane) {
  int l = lane & 15, hi = lane >> 4;
  v16bf b;
#pragma unroll
  for (int i = 0; i < 16; ++i) b[i] = src[kB_idx(i, hi) * ld + l];
  return b;
}
__device__ __forceinline__ v16bf load_bt_rm(const bf16* src, int ld, int lane) {
  int l = lane & 15, hi = lane >> 4;
  v16bf b;
#pragma unroll
  for (int i = 0; i < 16; ++i) b[i] = src[l * ld + kB_idx(i, hi)];
  return b;
}

// ---------------------------------------------------------------------------
// f32 -> bf16 conversion
// ---------------------------------------------------------------------------
__global__ void cvt_f32_bf16(const float* __restrict__ src, bf16* __restrict__ dst, long n) {
  long i = (long)blockIdx.x * blockDim.x + threadIdx.x;
  long stride = (long)gridDim.x * blockDim.x;
  for (; i < n; i += stride) dst[i] = (bf16)src[i];
}

// ---------------------------------------------------------------------------
// Tiled WMMA GEMM:  C[M x NC] = A[M x K](bf16) * W[NC x K]^T(bf16) + bias
// 128x128 workgroup tile, 8 waves (4x2), per-wave 32x64 (2x4 WMMA tiles).
// Double-buffered LDS; tile i+1 streamed by async-to-LDS DMA while WMMAs
// consume tile i.
// mode 0: QKV -> scatter bf16 into q/k/v [N,H,F,E]
// mode 1: FFN1 -> exact GELU -> bf16 row-major [M x NC]
// mode 2: FFN2 -> + resid  -> f32 row-major [M x MDL]
// ---------------------------------------------------------------------------
__global__ __launch_bounds__(256) void gemm_wmma_kernel(
    const bf16* __restrict__ A, const bf16* __restrict__ W, const float* __restrict__ bias,
    int K, int NC, int mode,
    bf16* __restrict__ qb, bf16* __restrict__ kb, bf16* __restrict__ vb,
    bf16* __restrict__ outb, const float* __restrict__ resid, float* __restrict__ outf)
{
  __shared__ bf16 s_a[2][128 * 32];   // [row][k]
  __shared__ bf16 s_b[2][128 * 32];   // [n][k]
  int t = threadIdx.x;
  int lane = t & 31, wid = t >> 5;
  int wm = wid & 3, wn = wid >> 2;
  int row0 = blockIdx.x * 128;
  int c0   = blockIdx.y * 128;

  v8f zero = {};
  v8f acc[2][4];
#pragma unroll
  for (int i = 0; i < 2; ++i)
#pragma unroll
    for (int j = 0; j < 4; ++j) acc[i][j] = zero;

  // per-thread async chunks: 512 x 16B chunks per tile, 2 per thread per matrix
  auto issue_tile = [&](int k0, int buf) {
#pragma unroll
    for (int c = 0; c < 2; ++c) {
      int ch = t + c * 256;
      int row = ch >> 2, seg = ch & 3;
      async_copy_b128(&s_a[buf][row * 32 + seg * 8],
                      A + (size_t)(row0 + row) * K + k0 + seg * 8);
      async_copy_b128(&s_b[buf][row * 32 + seg * 8],
                      W + (size_t)(c0 + row) * K + k0 + seg * 8);
    }
  };

  int nIter = K >> 5;
  issue_tile(0, 0);
  for (int it = 0; it < nIter; ++it) {
    wait_async0();
    __syncthreads();                       // tile `it` resident in buf it&1
    if (it + 1 < nIter) issue_tile((it + 1) << 5, (it + 1) & 1);
    const bf16* pa = s_a[it & 1];
    const bf16* pb = s_b[it & 1];
    v16bf af[2];
#pragma unroll
    for (int i = 0; i < 2; ++i)
      af[i] = load_a_bf16(pa + (wm * 32 + i * 16) * 32, 32, lane);
#pragma unroll
    for (int j = 0; j < 4; ++j) {
      v16bf bfrag = load_bt_rm(pb + (wn * 64 + j * 16) * 32, 32, lane);
#pragma unroll
      for (int i = 0; i < 2; ++i)
        acc[i][j] = wmma_bf16(af[i], bfrag, acc[i][j]);
    }
  }

  int nn = lane & 15, hh = lane >> 4;
#pragma unroll
  for (int i = 0; i < 2; ++i) {
#pragma unroll
    for (int j = 0; j < 4; ++j) {
      int c = c0 + wn * 64 + j * 16 + nn;
      float bb = bias[c];
#pragma unroll
      for (int r = 0; r < 8; ++r) {
        int grow = row0 + wm * 32 + i * 16 + r + 8 * hh;
        float val = acc[i][j][r] + bb;
        if (mode == 0) {
          int which = c >> 9, cc = c & 511;
          int h = cc >> 6, e = cc & 63;
          int f = grow >> 9, nb = grow & 511;       // N == 512
          bf16* dst = which == 0 ? qb : (which == 1 ? kb : vb);
          dst[(((size_t)nb * HEADS + h) * F_DIM + f) * E_DIM + e] = (bf16)val;
        } else if (mode == 1) {
          float g = 0.5f * val * (1.0f + erff(val * 0.70710678118654752f));
          outb[(size_t)grow * NC + c] = (bf16)g;
        } else {
          val += resid[(size_t)grow * MDL + c];
          outf[(size_t)grow * MDL + c] = val;
        }
      }
    }
  }
}

// ---------------------------------------------------------------------------
// Performer attention, one workgroup (4 waves) per (n, h).
// ---------------------------------------------------------------------------
__device__ __forceinline__ void dash_block(const bf16* s_x, const bf16* s_p, float* s_df,
                                           int lane, int w, float dn) {
  int nn = lane & 15, hh = lane >> 4;
#pragma unroll
  for (int i = 0; i < 2; ++i) {
    int l0 = w * 32 + i * 16;
    v16bf a0 = load_a_bf16(s_x + l0 * 64, 64, lane);
    v16bf a1 = load_a_bf16(s_x + l0 * 64 + 32, 64, lane);
#pragma unroll
    for (int j = 0; j < 8; ++j) {
      int m0 = j * 16;
      v16bf b0 = load_bt_rm(s_p + m0 * 64, 64, lane);
      v16bf b1 = load_bt_rm(s_p + m0 * 64 + 32, 64, lane);
      v8f acc = {};
      acc = wmma_bf16(a0, b0, acc);
      acc = wmma_bf16(a1, b1, acc);
#pragma unroll
      for (int r = 0; r < 8; ++r)
        s_df[(l0 + r + 8 * hh) * M_FEAT + m0 + nn] = acc[r] * dn;
    }
  }
}

__device__ __forceinline__ void feature_row(float* s_df, const bf16* s_x, int t,
                                            float dn, float ratio, float keps) {
  float mx = -1e30f;
  for (int m = 0; m < M_FEAT; ++m) mx = fmaxf(mx, s_df[t * M_FEAT + m]);
  float dg = 0.f;
  for (int e = 0; e < E_DIM; ++e) { float x = (float)s_x[t * E_DIM + e]; dg += x * x; }
  dg *= 0.5f * dn * dn;
  for (int m = 0; m < M_FEAT; ++m)
    s_df[t * M_FEAT + m] = ratio * (expf(s_df[t * M_FEAT + m] - dg - mx) + keps);
}

__global__ __launch_bounds__(128) void attention_kernel(
    const bf16* __restrict__ qb, const bf16* __restrict__ kb, const bf16* __restrict__ vb,
    const bf16* __restrict__ projb, float* __restrict__ tout)
{
  extern __shared__ char smem_raw[];
  float* s_df  = (float*)smem_raw;                 // [128][128] f32 (dash / features)
  bf16*  s_q   = (bf16*)(smem_raw + 65536);        // [128][64]
  bf16*  s_k   = s_q + 8192;
  bf16*  s_v   = s_k + 8192;
  bf16*  s_p   = s_v + 8192;
  bf16*  s_kv  = s_p + 8192;                       // [128 m][64 e] bf16
  float* s_ksum = (float*)(s_kv + 8192);           // [128]
  float* s_z    = s_ksum + 128;                    // [128]

  int t = threadIdx.x;
  int lane = t & 31, w = t >> 5;
  int nh = blockIdx.x;
  int nb = nh >> 3, h = nh & 7;
  size_t base = ((size_t)nb * HEADS + h) * (size_t)(F_DIM * E_DIM);

  { // stream q/k/v/proj tiles into LDS via async DMA (32 b128 per thread)
    const uint4* sq = (const uint4*)(qb + base);
    const uint4* sk = (const uint4*)(kb + base);
    const uint4* sv = (const uint4*)(vb + base);
    const uint4* sp = (const uint4*)projb;
    uint4* dq = (uint4*)s_q; uint4* dk = (uint4*)s_k;
    uint4* dv = (uint4*)s_v; uint4* dp = (uint4*)s_p;
    for (int i = t; i < 1024; i += 128) {
      async_copy_b128(dq + i, sq + i);
      async_copy_b128(dk + i, sk + i);
      async_copy_b128(dv + i, sv + i);
      async_copy_b128(dp + i, sp + i);
    }
    wait_async0();
  }
  __syncthreads();

  const float dn    = 0.35355339059327373f;   // 64^-0.25
  const float ratio = 0.08838834764831845f;   // 128^-0.5
  const float keps  = 1e-4f;

  // ---- K path: kdash -> kf (in s_df) -> ksum, kv ----
  dash_block(s_k, s_p, s_df, lane, w, dn);
  __syncthreads();
  feature_row(s_df, s_k, t, dn, ratio, keps);
  __syncthreads();
  { // ksum over s (column sums of kf)
    float s = 0.f;
    for (int ss = 0; ss < F_DIM; ++ss) s += s_df[ss * M_FEAT + t];
    s_ksum[t] = s;
  }
  { // kv[m][e] = sum_s kf[s][m] * v[s][e]
    int nn = lane & 15, hh = lane >> 4;
#pragma unroll
    for (int i = 0; i < 2; ++i) {
      int m0 = w * 32 + i * 16;
      v16bf af[4];
#pragma unroll
      for (int ks = 0; ks < 4; ++ks) af[ks] = load_at_f32(s_df + ks * 32 * M_FEAT + m0, M_FEAT, lane);
#pragma unroll
      for (int j = 0; j < 4; ++j) {
        int e0 = j * 16;
        v8f acc = {};
#pragma unroll
        for (int ks = 0; ks < 4; ++ks) {
          v16bf b = load_b_rm(s_v + ks * 32 * E_DIM + e0, E_DIM, lane);
          acc = wmma_bf16(af[ks], b, acc);
        }
#pragma unroll
        for (int r = 0; r < 8; ++r)
          s_kv[(m0 + r + 8 * hh) * E_DIM + e0 + nn] = (bf16)acc[r];
      }
    }
  }
  __syncthreads();

  // ---- Q path: qdash -> qf (reuse s_df) -> z -> t = qf @ kv * z ----
  dash_block(s_q, s_p, s_df, lane, w, dn);
  __syncthreads();
  feature_row(s_df, s_q, t, dn, ratio, keps);
  { // z[l] = 1 / (qf[l,:] . ksum + eps)
    float a = 0.f;
    for (int m = 0; m < M_FEAT; ++m) a += s_df[t * M_FEAT + m] * s_ksum[m];
    s_z[t] = 1.f / (a + 1e-6f);
  }
  __syncthreads();
  {
    int nn = lane & 15, hh = lane >> 4;
#pragma unroll
    for (int i = 0; i < 2; ++i) {
      int l0 = w * 32 + i * 16;
      v16bf af[4];
#pragma unroll
      for (int ks = 0; ks < 4; ++ks) af[ks] = load_a_f32(s_df + l0 * M_FEAT + ks * 32, M_FEAT, lane);
#pragma unroll
      for (int j = 0; j < 4; ++j) {
        int e0 = j * 16;
        v8f acc = {};
#pragma unroll
        for (int ks = 0; ks < 4; ++ks) {
          v16bf b = load_b_rm(s_kv + ks * 32 * E_DIM + e0, E_DIM, lane);
          acc = wmma_bf16(af[ks], b, acc);
        }
#pragma unroll
        for (int r = 0; r < 8; ++r) {
          int l = l0 + r + 8 * hh;   // frame index f
          size_t idx = ((size_t)l * N_DIM + nb) * MDL + h * E_DIM + e0 + nn;
          tout[idx] = acc[r] * s_z[l];
        }
      }
    }
  }
}

// ---------------------------------------------------------------------------
// Add + LayerNorm, one wave32 per row of 512, optional residual / bf16 copy
// ---------------------------------------------------------------------------
__global__ __launch_bounds__(256) void addnorm_kernel(
    const float* __restrict__ a, const float* __restrict__ resid,
    const float* __restrict__ g, const float* __restrict__ bta,
    float* __restrict__ outf, bf16* __restrict__ outb)
{
  int lane = threadIdx.x & 31, wid = threadIdx.x >> 5;
  size_t row = (size_t)blockIdx.x * 8 + wid;
  const float* pa = a + row * MDL;
  float v[16];
  float s = 0.f;
#pragma unroll
  for (int i = 0; i < 16; ++i) {
    int c = lane + i * 32;
    float x = pa[c];
    if (resid) x += resid[row * MDL + c];
    v[i] = x; s += x;
  }
#pragma unroll
  for (int off = 16; off > 0; off >>= 1) s += __shfl_xor(s, off, 32);
  float mean = s * (1.f / MDL);
  float vs = 0.f;
#pragma unroll
  for (int i = 0; i < 16; ++i) { float d = v[i] - mean; vs += d * d; }
#pragma unroll
  for (int off = 16; off > 0; off >>= 1) vs += __shfl_xor(vs, off, 32);
  float rstd = rsqrtf(vs * (1.f / MDL) + 1e-5f);
#pragma unroll
  for (int i = 0; i < 16; ++i) {
    int c = lane + i * 32;
    float o = (v[i] - mean) * rstd * g[c] + bta[c];
    outf[row * MDL + c] = o;
    if (outb) outb[row * MDL + c] = (bf16)o;
  }
}

// ---------------------------------------------------------------------------
extern "C" void kernel_launch(void* const* d_in, const int* in_sizes, int n_in,
                              void* d_out, int out_size, void* d_ws, size_t ws_size,
                              hipStream_t stream) {
  (void)in_sizes; (void)n_in; (void)out_size; (void)ws_size;
  const float* x     = (const float*)d_in[0];
  const float* w_qkv = (const float*)d_in[1];
  const float* b_qkv = (const float*)d_in[2];
  const float* proj  = (const float*)d_in[3];
  const float* ln1_g = (const float*)d_in[4];
  const float* ln1_b = (const float*)d_in[5];
  const float* ln2_g = (const float*)d_in[6];
  const float* ln2_b = (const float*)d_in[7];
  const float* w1    = (const float*)d_in[8];
  const float* b1    = (const float*)d_in[9];
  const float* w2    = (const float*)d_in[10];
  const float* b2    = (const float*)d_in[11];
  float* out = (float*)d_out;

  char* p = (char*)d_ws;
  auto carve = [&](size_t bytes) -> char* {
    char* r = p; p += (bytes + 255) & ~(size_t)255; return r;
  };
  bf16*  xb    = (bf16*)carve((size_t)ROWS * MDL * 2);
  bf16*  wqkvb = (bf16*)carve((size_t)3 * MDL * MDL * 2);
  bf16*  w1b   = (bf16*)carve((size_t)(MDL / 2) * MDL * 2);
  bf16*  w2b   = (bf16*)carve((size_t)MDL * (MDL / 2) * 2);
  bf16*  projb = (bf16*)carve((size_t)M_FEAT * E_DIM * 2);
  bf16*  qb    = (bf16*)carve((size_t)ROWS * MDL * 2);
  bf16*  kb    = (bf16*)carve((size_t)ROWS * MDL * 2);
  bf16*  vb    = (bf16*)carve((size_t)ROWS * MDL * 2);
  float* tbuf  = (float*)carve((size_t)ROWS * MDL * 4);   // also reused for FFN sum
  float* x1f   = (float*)carve((size_t)ROWS * MDL * 4);
  bf16*  x1b   = (bf16*)carve((size_t)ROWS * MDL * 2);
  bf16*  hb    = (bf16*)carve((size_t)ROWS * (MDL / 2) * 2);

  // 1) f32 -> bf16 conversions
  cvt_f32_bf16<<<1024, 256, 0, stream>>>(x, xb, (long)ROWS * MDL);
  cvt_f32_bf16<<<96, 256, 0, stream>>>(w_qkv, wqkvb, (long)3 * MDL * MDL);
  cvt_f32_bf16<<<16, 256, 0, stream>>>(w1, w1b, (long)(MDL / 2) * MDL);
  cvt_f32_bf16<<<16, 256, 0, stream>>>(w2, w2b, (long)MDL * (MDL / 2));
  cvt_f32_bf16<<<4, 256, 0, stream>>>(proj, projb, (long)M_FEAT * E_DIM);

  // 2) QKV projection (scatter into head layout)
  gemm_wmma_kernel<<<dim3(ROWS / 128, (3 * MDL) / 128), dim3(256), 0, stream>>>(
      xb, wqkvb, b_qkv, MDL, 3 * MDL, 0, qb, kb, vb, nullptr, nullptr, nullptr);

  // 3) Performer attention per (n, h); ~145 KB dynamic LDS
  size_t lds_bytes = 65536 + 5 * 16384 + 2 * 512;
  attention_kernel<<<dim3(N_DIM * HEADS), dim3(128), lds_bytes, stream>>>(
      qb, kb, vb, projb, tbuf);

  // 4) x1 = LN1(t + x); also bf16 copy for FFN
  addnorm_kernel<<<dim3(ROWS / 8), dim3(256), 0, stream>>>(
      tbuf, x, ln1_g, ln1_b, x1f, x1b);

  // 5) h = GELU(x1 @ w1^T + b1)
  gemm_wmma_kernel<<<dim3(ROWS / 128, (MDL / 2) / 128), dim3(256), 0, stream>>>(
      x1b, w1b, b1, MDL, MDL / 2, 1, nullptr, nullptr, nullptr, hb, nullptr, nullptr);

  // 6) s = h @ w2^T + b2 + x1   (reuse tbuf)
  gemm_wmma_kernel<<<dim3(ROWS / 128, MDL / 128), dim3(256), 0, stream>>>(
      hb, w2b, b2, MDL / 2, MDL, 2, nullptr, nullptr, nullptr, nullptr, x1f, tbuf);

  // 7) out = LN2(s)
  addnorm_kernel<<<dim3(ROWS / 8), dim3(256), 0, stream>>>(
      tbuf, nullptr, ln2_g, ln2_b, out, nullptr);
}
